// OumnixSimpleAI_1975684956474
// MI455X (gfx1250) — matrix operand order repaired and checked by reading, not compiled, not run on hardware
//
#include <hip/hip_runtime.h>
#include <hip/hip_bf16.h>
#include <hip/hip_fp16.h>
#include <stdint.h>

// ---------------------------------------------------------------------------
// MI455X (gfx1250, wave32) implementation of LoRA windowed attention.
// GEMMs + attention on v_wmma_f32_16x16x32_f16; GEMM A-tiles staged by the
// Tensor Data Mover (tensor_load_to_lds) with double-buffered LDS.
// Problem: B=2 N=2048 C=512 H=8 dh=64 WINDOW=64 R=8.
// ---------------------------------------------------------------------------

typedef __attribute__((ext_vector_type(16))) _Float16 v16h;
typedef __attribute__((ext_vector_type(8)))  float    v8f;
typedef __attribute__((ext_vector_type(4)))  unsigned int uint32x4;
typedef __attribute__((ext_vector_type(8)))  int          int32x8;
typedef __attribute__((ext_vector_type(4)))  int          int32x4;

#define HEADS   8
#define DH      64
#define NSEQ    2048
#define BATCH   2
#define CDIM    512
#define QKVDIM  1536
#define MROWS   (BATCH * NSEQ)     // 4096
#define LORA_R  8
#define LORA_SC (2.0f / 8.0f)

#if defined(__has_builtin)
#if __has_builtin(__builtin_amdgcn_tensor_load_to_lds) && \
    __has_builtin(__builtin_amdgcn_s_wait_tensorcnt)
#define HAVE_TDM 1
#endif
#endif
#ifndef HAVE_TDM
#define HAVE_TDM 0
#endif

union FragH { v16h h; uint32_t u[8]; };
union FragF { v8f  f; float    x[8]; };
union U4H   { uint4 q; _Float16 h[8]; };

// A-fragment k index (16-bit A, 16x32): VGPR pair table from ISA 7.12.2
__device__ __forceinline__ int kA(int v, int hi) {
    return ((v >> 2) << 4) + hi * 8 + ((v & 3) << 1);
}
// B-fragment k index (16-bit B, 32x16): same VGPR pair table as A
__device__ __forceinline__ int kB(int v, int hi) {
    return ((v >> 2) << 4) + hi * 8 + ((v & 3) << 1);
}

__device__ __forceinline__ v8f wmma_f16(v16h a, v16h b, v8f c) {
    return __builtin_amdgcn_wmma_f32_16x16x32_f16(
        false, a, false, b, (short)0, c, false, false);
}

// ---------------------------------------------------------------------------
// fp32 -> f16 conversion
// ---------------------------------------------------------------------------
__global__ __launch_bounds__(256) void cvt_f32_f16(
    const float* __restrict__ s, _Float16* __restrict__ d, int n) {
    int i = blockIdx.x * 256 + threadIdx.x;
    if (i < n) d[i] = (_Float16)s[i];
}

// ---------------------------------------------------------------------------
// LoRA down-projection: T[m][r] = sum_k X[m][k] * A[k][r]   (R=8, fp32 VALU)
// ---------------------------------------------------------------------------
__global__ __launch_bounds__(256) void lora_proj_f32(
    const float* __restrict__ X, const float* __restrict__ A,
    float* __restrict__ T, int M, int K) {
    int idx = blockIdx.x * 256 + threadIdx.x;
    if (idx >= M * LORA_R) return;
    int m = idx >> 3, r = idx & 7;
    float s = 0.f;
    for (int k = 0; k < K; ++k) s += X[(size_t)m * K + k] * A[k * LORA_R + r];
    T[idx] = s;
}

__global__ __launch_bounds__(256) void lora_proj_f16(
    const _Float16* __restrict__ X, const float* __restrict__ A,
    float* __restrict__ T, int M, int K) {
    int idx = blockIdx.x * 256 + threadIdx.x;
    if (idx >= M * LORA_R) return;
    int m = idx >> 3, r = idx & 7;
    float s = 0.f;
    for (int k = 0; k < K; ++k) s += (float)X[(size_t)m * K + k] * A[k * LORA_R + r];
    T[idx] = s;
}

// ---------------------------------------------------------------------------
// WMMA GEMM: out = A(MxK,f16) * Bw(KxN,f16) [+ bias] + LORA_SC * (loraT @ loraB)
// Block tile 64x128, BK=32, 256 threads = 8 waves, wave tile 16x64.
// Double-buffered LDS; A tile staged by the Tensor Data Mover (wave 0 issues,
// D# pad fields reproduce the KP=34 padded stride). B tile is transposed by
// hand (TDM cannot transpose).
// MODE 0: store f16 (qkv path).  MODE 1: store f32 + bias (output path).
// ---------------------------------------------------------------------------
#define BM 64
#define BN 128
#define BK 32
#define KP 34   // padded LDS K stride (f16 units) = 17 DWORDs

#if HAVE_TDM
// Issue one TDM 2-D tile load: BMxBK f16 tile, row stride = gstride elements,
// LDS rows padded to 17 DWORDs via pad_interval=16DW (code 3), pad_amount=1DW.
__device__ __forceinline__ void tdm_stage_tile(
    const _Float16* gsrc, uint32_t lds_byte_addr, unsigned gstride) {
    uint64_t ga = (uint64_t)(size_t)gsrc;
    uint32x4 g0;
    g0[0] = 1u;                                              // count=1, user D#
    g0[1] = lds_byte_addr;                                   // lds_addr
    g0[2] = (unsigned)(ga & 0xFFFFFFFFu);                    // global_addr lo
    g0[3] = (unsigned)((ga >> 32) & 0x01FFFFFFu) | (2u << 30); // addr hi | type=2
    int32x8 g1;
    g1[0] = (1 << 16) | (1 << 20) | (3 << 22);  // data_size=2B, pad_en, intvl=16DW, amt=1DW
    g1[1] = (int)((gstride & 0xFFFFu) << 16);   // abar=0 | tensor_dim0[15:0]
    g1[2] = (int)(((gstride >> 16) & 0xFFFFu) | ((MROWS & 0xFFFFu) << 16)); // dim0 hi | dim1 lo
    g1[3] = (int)(BK << 16);                    // tensor_dim1 hi=0 | tile_dim0=32
    g1[4] = BM;                                 // tile_dim1=64 | tile_dim2=0
    g1[5] = (int)gstride;                       // tensor_dim0_stride lo
    g1[6] = 0;                                  // stride0 hi | stride1 lo
    g1[7] = 0;
    int32x4 gz4 = {0, 0, 0, 0};
    int32x8 gz8 = {0, 0, 0, 0, 0, 0, 0, 0};
    // 6-arg form (clang-23 / therock-10.0 headers): groups 2/3/4 zero-filled.
    __builtin_amdgcn_tensor_load_to_lds(g0, g1, gz4, gz4, gz8, 0);
}
#endif

template<int MODE>
__global__ __launch_bounds__(256) void gemm_wmma(
    const _Float16* __restrict__ A, const _Float16* __restrict__ Bw,
    int M, int N, int K,
    const float* __restrict__ loraT,  // M x 8
    const float* __restrict__ loraB,  // 8 x N
    const float* __restrict__ bias,   // N (MODE 1)
    _Float16* __restrict__ outh,      // MODE 0
    float* __restrict__ outf)         // MODE 1
{
    __shared__ _Float16 As[2][BM][KP];
    __shared__ _Float16 Bt[2][BN][KP];   // transposed B tile: Bt[buf][n][k]

    const int tid  = threadIdx.x;
    const int lane = tid & 31, wave = tid >> 5;
    const int wm = wave & 3, wn = wave >> 2;        // wave tile: rows wm*16, cols wn*64
    const int hi = lane >> 4, ln = lane & 15;
    const int m0 = blockIdx.x * BM, n0 = blockIdx.y * BN;

    // Stage A tile for K-offset kk into buffer buf
    auto stageA = [&](int buf, int kk) {
#if HAVE_TDM
        if (wave == 0)
            tdm_stage_tile(A + (size_t)m0 * K + kk,
                           (uint32_t)(size_t)&As[buf][0][0], (unsigned)K);
#else
        int r = tid >> 2, cg = (tid & 3) << 3;
        uint4 d = *(const uint4*)(A + (size_t)(m0 + r) * K + kk + cg);
        uint32_t* dst = (uint32_t*)&As[buf][r][cg];
        dst[0] = d.x; dst[1] = d.y; dst[2] = d.z; dst[3] = d.w;
#endif
    };
    // Stage B tile (transposed) for K-offset kk into buffer buf
    auto stageB = [&](int buf, int kk) {
        int r = tid >> 3, cg = (tid & 7) << 4;
        const uint4* src = (const uint4*)(Bw + (size_t)(kk + r) * N + n0 + cg);
        U4H u0, u1; u0.q = src[0]; u1.q = src[1];
#pragma unroll
        for (int i = 0; i < 8; ++i) Bt[buf][cg + i][r]     = u0.h[i];
#pragma unroll
        for (int i = 0; i < 8; ++i) Bt[buf][cg + 8 + i][r] = u1.h[i];
    };

    v8f acc[4] = {};
    const int niter = K / BK;

    stageA(0, 0);
    stageB(0, 0);

    for (int it = 0; it < niter; ++it) {
        const int cur = it & 1;
#if HAVE_TDM
        if (wave == 0) __builtin_amdgcn_s_wait_tensorcnt((short)0);
#endif
        __syncthreads();   // publishes TDM A(cur) + manual B(cur) stores

        FragH af;
#pragma unroll
        for (int v = 0; v < 8; ++v)
            af.u[v] = *(const uint32_t*)&As[cur][wm * 16 + ln][kA(v, hi)];
#pragma unroll
        for (int s = 0; s < 4; ++s) {
            FragH bf;
#pragma unroll
            for (int v = 0; v < 8; ++v)
                bf.u[v] = *(const uint32_t*)&Bt[cur][wn * 64 + s * 16 + ln][kB(v, hi)];
            acc[s] = wmma_f16(af.h, bf.h, acc[s]);
        }

        // Prefetch next tile into the other buffer, overlapped with WMMA work.
        if (it + 1 < niter) {
            stageA(cur ^ 1, (it + 1) * BK);
            stageB(cur ^ 1, (it + 1) * BK);
        }
    }

    // Epilogue: LoRA rank-8 correction (+bias for MODE 1)
    __builtin_prefetch(&loraB[n0 + wn * 64 + ln], 0, 1);
#pragma unroll
    for (int v = 0; v < 8; ++v) {
        const int m = m0 + wm * 16 + v + 8 * hi;
        float la[LORA_R];
#pragma unroll
        for (int r = 0; r < LORA_R; ++r) la[r] = loraT[(size_t)m * LORA_R + r];
        FragF* accu = (FragF*)acc;
#pragma unroll
        for (int s = 0; s < 4; ++s) {
            const int n = n0 + wn * 64 + s * 16 + ln;
            float lv = 0.f;
#pragma unroll
            for (int r = 0; r < LORA_R; ++r) lv += la[r] * loraB[(size_t)r * N + n];
            float val = accu[s].x[v] + LORA_SC * lv;
            if (MODE == 0) {
                outh[(size_t)m * N + n] = (_Float16)val;
            } else {
                outf[(size_t)m * N + n] = val + bias[n];
            }
        }
    }
}

// ---------------------------------------------------------------------------
// Attention: one wave per (b, head, 16-query tile).
// Local: 5 key tiles covering [q0-64, q0+15], causal+window mask, softmax.
// Global: 32 strided keys (::64), softmax, same output fragments.
// ---------------------------------------------------------------------------
__global__ __launch_bounds__(128) void attn_wmma(
    const _Float16* __restrict__ qkv,   // (B*N) x 1536 f16
    _Float16* __restrict__ outh)        // (B*N) x 512  f16
{
    __shared__ _Float16 Pl[4][16][96];  // per-wave P matrix (A-layout staging)
    __shared__ _Float16 Vt[4][64][KP];  // per-wave transposed V chunk

    const int lane = threadIdx.x & 31;
    const int wv   = threadIdx.x >> 5;
    const int gw   = blockIdx.x * 4 + wv;       // 2048 waves total
    const int qtile = gw & 127;
    const int hd    = (gw >> 7) & 7;
    const int b     = gw >> 10;
    const int hi = lane >> 4, ln = lane & 15;
    const int q0 = qtile * 16;
    const int h64 = hd * DH;
    const float scale = 0.125f;   // dh^-0.5
    const _Float16* base = qkv + (size_t)b * NSEQ * QKVDIM;

    // Q fragments: two K=32 chunks over dh=64
    v16h qf[2];
#pragma unroll
    for (int c = 0; c < 2; ++c) {
        FragH f;
#pragma unroll
        for (int v = 0; v < 8; ++v)
            f.u[v] = *(const uint32_t*)(base + (size_t)(q0 + ln) * QKVDIM +
                                        h64 + c * 32 + kA(v, hi));
        qf[c] = f.h;
    }

    // ---- local scores: 5 key tiles (S = Q K^T) ----
    FragF sc[5];
#pragma unroll
    for (int t = 0; t < 5; ++t) {
        int kt = qtile - 4 + t;
        int krow = (kt < 0 ? 0 : kt) * 16;      // clamp; masked below
        v8f a = {};
#pragma unroll
        for (int c = 0; c < 2; ++c) {
            FragH bf;
#pragma unroll
            for (int v = 0; v < 8; ++v)
                bf.u[v] = *(const uint32_t*)(base + (size_t)(krow + ln) * QKVDIM +
                                             CDIM + h64 + c * 32 + kB(v, hi));
            a = wmma_f16(qf[c], bf.h, a);
        }
        sc[t].f = a;
    }

    // ---- mask + row softmax (rows live across 16-lane halves) ----
#pragma unroll
    for (int v = 0; v < 8; ++v) {
        const int i = q0 + v + 8 * hi;
        float mx = -3.0e38f;
#pragma unroll
        for (int t = 0; t < 5; ++t) {
            int kt = qtile - 4 + t;
            int j = kt * 16 + ln;
            bool valid = (kt >= 0) && (j <= i) && ((i - j) <= 64);
            float s = valid ? sc[t].x[v] * scale : -3.0e38f;
            sc[t].x[v] = s;
            mx = fmaxf(mx, s);
        }
#pragma unroll
        for (int off = 8; off > 0; off >>= 1) mx = fmaxf(mx, __shfl_xor(mx, off, 32));
        float sum = 0.f;
#pragma unroll
        for (int t = 0; t < 5; ++t) {
            float p = __expf(sc[t].x[v] - mx);
            sc[t].x[v] = p;
            sum += p;
        }
#pragma unroll
        for (int off = 8; off > 0; off >>= 1) sum += __shfl_xor(sum, off, 32);
        float inv = 1.0f / sum;
#pragma unroll
        for (int t = 0; t < 5; ++t)
            Pl[wv][v + 8 * hi][t * 16 + ln] = (_Float16)(sc[t].x[v] * inv);
        Pl[wv][v + 8 * hi][80 + ln] = (_Float16)0.f;   // zero pad tile
    }

    // ---- out_local = P V : 3 chunks of 32 keys ----
    v8f oacc[4] = {};
#pragma unroll
    for (int c = 0; c < 3; ++c) {
        int gkey = (qtile - 4) * 16 + c * 32 + lane;
        gkey = gkey < 0 ? 0 : (gkey > NSEQ - 1 ? NSEQ - 1 : gkey);
        const uint4* vsrc = (const uint4*)(base + (size_t)gkey * QKVDIM + 2 * CDIM + h64);
#pragma unroll
        for (int g = 0; g < 4; ++g) {
            U4H u; u.q = vsrc[g];
#pragma unroll
            for (int d = 0; d < 8; ++d) Vt[wv][g * 8 + d][lane] = u.h[d];
        }
        FragH pf;
#pragma unroll
        for (int v = 0; v < 8; ++v)
            pf.u[v] = *(const uint32_t*)&Pl[wv][ln][c * 32 + kA(v, hi)];
#pragma unroll
        for (int s = 0; s < 4; ++s) {
            FragH bf;
#pragma unroll
            for (int v = 0; v < 8; ++v)
                bf.u[v] = *(const uint32_t*)&Vt[wv][s * 16 + ln][kB(v, hi)];
            oacc[s] = wmma_f16(pf.h, bf.h, oacc[s]);
        }
    }

    // ---- global attention: 32 strided keys (rows g*64), no mask ----
    FragF sg[2];
#pragma unroll
    for (int t = 0; t < 2; ++t) {
        v8f a = {};
#pragma unroll
        for (int c = 0; c < 2; ++c) {
            FragH bf;
#pragma unroll
            for (int v = 0; v < 8; ++v)
                bf.u[v] = *(const uint32_t*)(base + (size_t)(t * 16 + ln) * 64 * QKVDIM +
                                             CDIM + h64 + c * 32 + kB(v, hi));
            a = wmma_f16(qf[c], bf.h, a);
        }
        sg[t].f = a;
    }
#pragma unroll
    for (int v = 0; v < 8; ++v) {
        float s0 = sg[0].x[v] * scale, s1 = sg[1].x[v] * scale;
        float mx = fmaxf(s0, s1);
#pragma unroll
        for (int off = 8; off > 0; off >>= 1) mx = fmaxf(mx, __shfl_xor(mx, off, 32));
        float p0 = __expf(s0 - mx), p1 = __expf(s1 - mx);
        float sum = p0 + p1;
#pragma unroll
        for (int off = 8; off > 0; off >>= 1) sum += __shfl_xor(sum, off, 32);
        float inv = 1.0f / sum;
        Pl[wv][v + 8 * hi][ln]      = (_Float16)(p0 * inv);
        Pl[wv][v + 8 * hi][16 + ln] = (_Float16)(p1 * inv);
    }
    {
        const uint4* vsrc = (const uint4*)(base + (size_t)lane * 64 * QKVDIM + 2 * CDIM + h64);
#pragma unroll
        for (int g = 0; g < 4; ++g) {
            U4H u; u.q = vsrc[g];
#pragma unroll
            for (int d = 0; d < 8; ++d) Vt[wv][g * 8 + d][lane] = u.h[d];
        }
        FragH pf;
#pragma unroll
        for (int v = 0; v < 8; ++v)
            pf.u[v] = *(const uint32_t*)&Pl[wv][ln][kA(v, hi)];
#pragma unroll
        for (int s = 0; s < 4; ++s) {
            FragH bf;
#pragma unroll
            for (int v = 0; v < 8; ++v)
                bf.u[v] = *(const uint32_t*)&Vt[wv][s * 16 + ln][kB(v, hi)];
            oacc[s] = wmma_f16(pf.h, bf.h, oacc[s]);
        }
    }

    // ---- store out tile (b, n, h*64 + d) as f16 ----
    FragF* oa = (FragF*)oacc;
#pragma unroll
    for (int s = 0; s < 4; ++s)
#pragma unroll
        for (int v = 0; v < 8; ++v) {
            int m = q0 + v + 8 * hi;
            int d = h64 + s * 16 + ln;
            outh[((size_t)b * NSEQ + m) * CDIM + d] = (_Float16)oa[s].x[v];
        }
}

// ---------------------------------------------------------------------------
// Launch
// ---------------------------------------------------------------------------
extern "C" void kernel_launch(void* const* d_in, const int* in_sizes, int n_in,
                              void* d_out, int out_size, void* d_ws, size_t ws_size,
                              hipStream_t stream) {
    const float* x       = (const float*)d_in[0];
    const float* w_qkv   = (const float*)d_in[1];
    const float* lA_qkv  = (const float*)d_in[2];
    const float* lB_qkv  = (const float*)d_in[3];
    const float* w_out   = (const float*)d_in[4];
    const float* b_out   = (const float*)d_in[5];
    const float* lA_out  = (const float*)d_in[6];
    const float* lB_out  = (const float*)d_in[7];
    float* outp = (float*)d_out;

    char* ws = (char*)d_ws;
    size_t off = 0;
    auto alloc = [&](size_t bytes) -> void* {
        off = (off + 255) & ~(size_t)255;
        void* p = ws + off;
        off += bytes;
        return p;
    };
    _Float16* xh    = (_Float16*)alloc((size_t)MROWS * CDIM   * 2);
    _Float16* wqkvh = (_Float16*)alloc((size_t)CDIM  * QKVDIM * 2);
    _Float16* wouth = (_Float16*)alloc((size_t)CDIM  * CDIM   * 2);
    _Float16* qkvh  = (_Float16*)alloc((size_t)MROWS * QKVDIM * 2);
    _Float16* outh  = (_Float16*)alloc((size_t)MROWS * CDIM   * 2);
    float*    xa    = (float*)   alloc((size_t)MROWS * LORA_R * 4);
    float*    oa    = (float*)   alloc((size_t)MROWS * LORA_R * 4);
    if (off > ws_size) return;   // workspace too small (≈23.5 MB needed)

    const int nx = MROWS * CDIM, nwq = CDIM * QKVDIM, nwo = CDIM * CDIM;
    cvt_f32_f16<<<(nx  + 255) / 256, 256, 0, stream>>>(x,     xh,    nx);
    cvt_f32_f16<<<(nwq + 255) / 256, 256, 0, stream>>>(w_qkv, wqkvh, nwq);
    cvt_f32_f16<<<(nwo + 255) / 256, 256, 0, stream>>>(w_out, wouth, nwo);

    lora_proj_f32<<<(MROWS * LORA_R + 255) / 256, 256, 0, stream>>>(
        x, lA_qkv, xa, MROWS, CDIM);

    gemm_wmma<0><<<dim3(MROWS / BM, QKVDIM / BN), 256, 0, stream>>>(
        xh, wqkvh, MROWS, QKVDIM, CDIM, xa, lB_qkv, nullptr, qkvh, nullptr);

    attn_wmma<<<(BATCH * HEADS * (NSEQ / 16)) / 4, 128, 0, stream>>>(qkvh, outh);

    lora_proj_f16<<<(MROWS * LORA_R + 255) / 256, 256, 0, stream>>>(
        outh, lA_out, oa, MROWS, CDIM);

    gemm_wmma<1><<<dim3(MROWS / BM, CDIM / BN), 256, 0, stream>>>(
        outh, wouth, MROWS, CDIM, CDIM, oa, lB_out, b_out, nullptr, outp);
}
